// OriginalImplementationHistogramLoss_27814208209512
// MI455X (gfx1250) — matrix operand order repaired and checked by reading, not compiled, not run on hardware
//
#include <hip/hip_runtime.h>
#include <hip/hip_bf16.h>

typedef __attribute__((ext_vector_type(16))) _Float16 v16h;
typedef __attribute__((ext_vector_type(8)))  float    v8f;

#define N_FEAT 4096
#define D_FEAT 128
#define TS     101                    // histogram nodes
#define STEP_INV 50.0f                // 1 / (2/100)
#define NBLK   (N_FEAT / 128)         // 32 macro-tile rows/cols
#define NTRI   (NBLK * (NBLK + 1) / 2)// 528 upper-triangular 128x128 macro tiles
#define HPAD   104                    // padded per-histogram stride in LDS

union FragU { v16h v; float4 q[2]; };

// ---------------------------------------------------------------- f32 -> f16
__global__ __launch_bounds__(256) void cvt_kernel(const float* __restrict__ f,
                                                  _Float16* __restrict__ h) {
  int i = blockIdx.x * blockDim.x + threadIdx.x;
  h[i] = (_Float16)f[i];
}

// -------------------------------------------------- zero global histograms
__global__ __launch_bounds__(256) void zero_hist_kernel(float* __restrict__ g) {
  int i = threadIdx.x;
  if (i < 2 * TS) g[i] = 0.0f;
}

// histogram update for one similarity value (triangular-kernel binning)
__device__ __forceinline__ void hist_update(float* hw, int off, float s) {
  const float pos = (s + 1.0f) * STEP_INV;
  int idx = (int)floorf(pos);
  idx = idx < 0 ? 0 : (idx > TS - 1 ? TS - 1 : idx);
  const float frac = pos - (float)idx;
  const int idx2 = (idx + 1 > TS - 1) ? TS - 1 : idx + 1;
  atomicAdd(&hw[off + idx],  1.0f - frac);
  atomicAdd(&hw[off + idx2], frac);
}

// --------------------------------------- fused WMMA GEMM + soft histogram
__global__ __launch_bounds__(256) void gemm_hist_kernel(
    const _Float16* __restrict__ feat,   // [N_FEAT, D_FEAT] row-major f16
    const int* __restrict__ classes,     // [N_FEAT]
    float* __restrict__ ghist) {         // [2*TS]: pos then neg
  __shared__ float whist[8][2 * HPAD];   // per-wave privatized pos/neg hists
  __shared__ int clsA[128];
  __shared__ int clsB[128];

  const int tid  = threadIdx.x;
  const int wave = tid >> 5;
  const int lane = tid & 31;

  for (int i = tid; i < 8 * 2 * HPAD; i += 256) (&whist[0][0])[i] = 0.0f;

  // decode upper-triangular macro-tile (bi <= bj)
  int t = blockIdx.x;
  int bi = 0;
  while (t >= (NBLK - bi)) { t -= (NBLK - bi); ++bi; }
  const int bj = bi + t;
  const bool diag = (bi == bj);

  if (tid < 128) clsA[tid] = classes[bi * 128 + tid];
  else           clsB[tid - 128] = classes[bj * 128 + (tid - 128)];
  __syncthreads();

  const int lh = lane >> 4;   // half-wave index (0/1)
  const int ll = lane & 15;

  // A fragments: this wave's 16 rows of F, all 4 K-chunks, kept in VGPRs.
  // 16-bit A 16x32 layout: lane<16 -> M=ll, K={kc*32+0..7, kc*32+16..23};
  // lane>=16 -> same M, K ranges shifted by 8.
  const _Float16* arow = feat + (size_t)(bi * 128 + wave * 16 + ll) * D_FEAT;
  v16h afr[4];
#pragma unroll
  for (int kc = 0; kc < 4; ++kc) {
    FragU a;
    const _Float16* p = arow + kc * 32 + lh * 8;
    a.q[0] = *(const float4*)(p);        // elems 0..7
    a.q[1] = *(const float4*)(p + 16);   // elems 8..15
    afr[kc] = a.v;
  }

  // nt-invariant: the 8 row-class values this lane's C elements use
  int ci[8];
#pragma unroll
  for (int r = 0; r < 8; ++r) ci[r] = clsA[wave * 16 + 8 * lh + r];

  float* const hw = whist[wave];

  // Diagonal blocks: sub-tiles with nt < wave are strictly below the
  // diagonal -> skip entirely. Only nt == wave needs per-lane masking.
  const int nt0 = diag ? wave : 0;
#pragma unroll 1
  for (int nt = nt0; nt < 8; ++nt) {
    // B fragment: B = F^T, so column n of B is row n of F; same K layout as A
    // (lane<16 -> K {0..7,16..23}, lane>=16 -> K {8..15,24..31}).
    const _Float16* brow = feat + (size_t)(bj * 128 + nt * 16 + ll) * D_FEAT;
    FragU b[4];
#pragma unroll
    for (int kc = 0; kc < 4; ++kc) {
      const _Float16* p = brow + kc * 32 + lh * 8;
      b[kc].q[0] = *(const float4*)(p);
      b[kc].q[1] = *(const float4*)(p + 16);
    }
    v8f acc = {};
#pragma unroll
    for (int kc = 0; kc < 4; ++kc)
      acc = __builtin_amdgcn_wmma_f32_16x16x32_f16(
          false, afr[kc], false, b[kc].v, (short)0, acc, false, false);

    // C/D layout: element r of acc, lane -> M = r + 8*lh, N = ll.
    const int cj = clsB[nt * 16 + ll];

    if (diag && nt == wave) {
      // partially masked tile: keep only (8*lh + r) < ll within the tile
#pragma unroll
      for (int r = 0; r < 8; ++r) {
        if (8 * lh + r < ll)
          hist_update(hw, (ci[r] == cj) ? 0 : HPAD, acc[r]);
      }
    } else {
      // fully valid tile: branch-free
#pragma unroll
      for (int r = 0; r < 8; ++r)
        hist_update(hw, (ci[r] == cj) ? 0 : HPAD, acc[r]);
    }
  }
  __syncthreads();

  // reduce 8 per-wave hists -> global (weight sums = histogram sums)
  for (int i = tid; i < 2 * TS; i += 256) {
    const int isNeg = (i >= TS);
    const int bbin = isNeg ? i - TS : i;
    const int off = isNeg ? HPAD : 0;
    float s = 0.0f;
#pragma unroll
    for (int w = 0; w < 8; ++w) s += whist[w][off + bbin];
    atomicAdd(&ghist[i], s);
  }
}

// --------------------------------- normalize, cumsum, dot -> scalar loss
__global__ __launch_bounds__(128) void finalize_kernel(
    const float* __restrict__ ghist, float* __restrict__ out) {
  __shared__ float hp[TS];
  __shared__ float hn[TS];
  const int tid = threadIdx.x;
  if (tid < TS) { hp[tid] = ghist[tid]; hn[tid] = ghist[TS + tid]; }
  __syncthreads();
  if (tid == 0) {
    float sp = 0.0f, sn = 0.0f;
    for (int i = 0; i < TS; ++i) { sp += hp[i]; sn += hn[i]; }
    const float invp = sp > 0.0f ? 1.0f / sp : 0.0f;
    const float invn = sn > 0.0f ? 1.0f / sn : 0.0f;
    float cdf = 0.0f, loss = 0.0f;
    for (int i = 0; i < TS; ++i) {
      cdf  += hp[i] * invp;          // inclusive cumsum of normalized pos
      loss += hn[i] * invn * cdf;
    }
    out[0] = loss;
  }
}

extern "C" void kernel_launch(void* const* d_in, const int* in_sizes, int n_in,
                              void* d_out, int out_size, void* d_ws, size_t ws_size,
                              hipStream_t stream) {
  (void)in_sizes; (void)n_in; (void)out_size; (void)ws_size;
  const float* feats   = (const float*)d_in[0];
  const int*   classes = (const int*)d_in[1];

  _Float16* f16   = (_Float16*)d_ws;
  float*    ghist = (float*)((char*)d_ws + (size_t)N_FEAT * D_FEAT * sizeof(_Float16));
  float*    out   = (float*)d_out;

  cvt_kernel<<<(N_FEAT * D_FEAT) / 256, 256, 0, stream>>>(feats, f16);
  zero_hist_kernel<<<1, 256, 0, stream>>>(ghist);
  gemm_hist_kernel<<<NTRI, 256, 0, stream>>>(f16, classes, ghist);
  finalize_kernel<<<1, 128, 0, stream>>>(ghist, out);
}
